// EgoActorCritic_55482387530362
// MI455X (gfx1250) — matrix operand order repaired and checked by reading, not compiled, not run on hardware
//
#include <hip/hip_runtime.h>
#include <hip/hip_bf16.h>
#include <math.h>

typedef __attribute__((ext_vector_type(16))) _Float16 v16h;
typedef __attribute__((ext_vector_type(8)))  float    v8f;

typedef __attribute__((address_space(3))) float    lds_f32;
typedef __attribute__((address_space(3))) _Float16 lds_f16;

#define R_GRAPHS 64
#define N_MAX    4096
#define IN_DIM   64
#define HIDDEN   128
#define K_MAX    512
#define NEGV     (-1e9f)

// ---------------- workspace layout (bytes) ----------------
// [0      .. 16384)  fragA : 8192 f16  (8 nt * 2 kf * 32 lanes * 16 halves)
// [16384  .. 32768)  fragC : 8192 f16  (contiguous with fragA -- relied upon)
// [32768  .. 65536)  E_sum : 64*128 f32
// [65536  .. 65792)  count : 64 f32
// [65792  .. +1MB )  scores: 64*4096 f32

// async copy: one wave32 instruction moves 512B (16B/lane), offset applies to
// BOTH global and LDS addresses (ISA 15.18.3).
#define ASYNC_B128(ldsp, gaddr, OFF)                                     \
    asm volatile("global_load_async_to_lds_b128 %0, %1, off offset:" OFF \
                 :: "v"(ldsp), "v"(gaddr) : "memory")

#define WAIT_ASYNC(N) asm volatile("s_wait_asynccnt " N ::: "memory")

// copy 4KB (gbase/lbase already include the per-lane +lane*16B part);
// LDS rows padded 256B->272B, so LDS base advances an extra 32B per step.
__device__ __forceinline__ void async_copy_tile_padded(unsigned long long ga,
                                                       lds_f32* lbase) {
    ASYNC_B128(lbase + 0,  ga, "0");
    ASYNC_B128(lbase + 8,  ga, "512");
    ASYNC_B128(lbase + 16, ga, "1024");
    ASYNC_B128(lbase + 24, ga, "1536");
    ASYNC_B128(lbase + 32, ga, "2048");
    ASYNC_B128(lbase + 40, ga, "2560");
    ASYNC_B128(lbase + 48, ga, "3072");
    ASYNC_B128(lbase + 56, ga, "3584");
}

__device__ __forceinline__ void async_copy_4k(unsigned long long ga,
                                              lds_f16* lp) {
    ASYNC_B128(lp, ga, "0");
    ASYNC_B128(lp, ga, "512");
    ASYNC_B128(lp, ga, "1024");
    ASYNC_B128(lp, ga, "1536");
    ASYNC_B128(lp, ga, "2048");
    ASYNC_B128(lp, ga, "2560");
    ASYNC_B128(lp, ga, "3072");
    ASYNC_B128(lp, ga, "3584");
}

// =====================================================================
// Kernel 1: pack W_a / W_c into WMMA B-fragment order (f16) + zero E_sum
// B (32x16, f16): lane n0=lane&15 is column; lanes<16 hold K=0..7,16..23,
// lanes>=16 hold K=8..15,24..31 (within a K-fragment of 32). 16 halves per lane,
// stored contiguously: frag f = nt*2+kf, offset f*512 + lane*16 + j.
// =====================================================================
__global__ __launch_bounds__(256) void prep_kernel(
    const float* __restrict__ W_a, const float* __restrict__ W_c,
    _Float16* __restrict__ fragA, _Float16* __restrict__ fragC,
    float* __restrict__ E_sum)
{
    int gtid = blockIdx.x * 256 + threadIdx.x;   // grid = 64 blocks -> 16384
    if (gtid < 16384) {
        int mat  = gtid >> 13;          // 0 = W_a, 1 = W_c
        int rem  = gtid & 8191;
        int f    = rem >> 9;            // fragment id 0..15  (nt*2+kf)
        int lane = (rem >> 4) & 31;
        int j    = rem & 15;
        int nt   = f >> 1;
        int kf   = f & 1;
        int k = kf * 32 + ((lane >= 16) ? 8 : 0) + ((j >= 8) ? 8 : 0) + j;   // 0..63
        int n = nt * 16 + (lane & 15);                   // 0..127
        const float* W  = mat ? W_c : W_a;
        _Float16*    dst = mat ? fragC : fragA;
        dst[rem] = (_Float16)W[k * HIDDEN + n];
    }
    if (gtid < R_GRAPHS * HIDDEN) E_sum[gtid] = 0.0f;
}

// =====================================================================
// Kernel 2: per-graph valid-node counts
// =====================================================================
__global__ __launch_bounds__(256) void count_kernel(
    const int* __restrict__ node_mask, float* __restrict__ cnt)
{
    int r = blockIdx.x;
    __shared__ int s[256];
    int acc = 0;
    for (int i = threadIdx.x; i < N_MAX; i += 256)
        acc += node_mask[r * N_MAX + i];
    s[threadIdx.x] = acc;
    __syncthreads();
    for (int off = 128; off > 0; off >>= 1) {
        if (threadIdx.x < off) s[threadIdx.x] += s[threadIdx.x + off];
        __syncthreads();
    }
    if (threadIdx.x == 0) cnt[r] = (float)s[0];
}

// =====================================================================
// Kernel 3: fused encode. grid = (8, 64), block = 256 (8 waves).
// x tiles double-buffered in LDS via global_load_async_to_lds_b128;
// weight fragments staged in LDS once per block, reloaded per tile through
// an opaque pointer (keeps VGPR pressure low).
// =====================================================================
__global__ __launch_bounds__(256) void encode_kernel(
    const float* __restrict__ x, const int* __restrict__ node_mask,
    const float* __restrict__ b_a, const float* __restrict__ b_c,
    const float* __restrict__ ln_g, const float* __restrict__ ln_b,
    const float* __restrict__ w_head, const float* __restrict__ b_head,
    const _Float16* __restrict__ fragA,   // fragC = fragA + 8192 (contiguous)
    float* __restrict__ E_sum, float* __restrict__ scores)
{
    const int g    = blockIdx.y;
    const int lane = threadIdx.x & 31;
    const int wave = threadIdx.x >> 5;
    const int m0   = lane & 15;
    const int half = lane >> 4;

    // LDS: fragments 32KB + x double-buffers 8*2*4352B + critic reduce
    __shared__ __attribute__((aligned(32))) _Float16 sfrag[16384];
    __shared__ __attribute__((aligned(16))) float    sx[8 * 2 * 1088];
    __shared__ float lds_c[HIDDEN];
    if (threadIdx.x < HIDDEN) lds_c[threadIdx.x] = 0.0f;

    // ---- stage weight fragments (32KB, contiguous fragA|fragC) ----
    {
        unsigned long long ga =
            (unsigned long long)((const char*)fragA + wave * 4096 + lane * 16);
        lds_f16* lp = (lds_f16*)&sfrag[wave * 2048 + lane * 8];
        async_copy_4k(ga, lp);                       // ASYNCcnt = 8
    }

    const int node_base = blockIdx.x * 512 + wave * 64;  // this wave's 4 tiles
    const unsigned long long gx0 =
        (unsigned long long)(const char*)(x + ((size_t)g * N_MAX + node_base) * IN_DIM)
        + (unsigned)lane * 16u;
    lds_f32* lx0 = (lds_f32*)&sx[wave * 2176 + half * 68 + m0 * 4];

    async_copy_tile_padded(gx0, lx0);                // tile 0 -> buf 0, cnt=16
    WAIT_ASYNC("0x8");                               // fragment copies done
    __syncthreads();                                 // frags + lds_c visible

    // per-lane per-nt parameters (column n = nt*16 + m0, constant over tiles)
    float gv[8], bv[8], wv[8], bav[8], bcv[8];
#pragma unroll
    for (int nt = 0; nt < 8; ++nt) {
        int n   = nt * 16 + m0;
        gv[nt]  = ln_g[n];
        bv[nt]  = ln_b[n];
        wv[nt]  = w_head[n];
        bav[nt] = b_a[n];
        bcv[nt] = b_c[n];
    }
    const float bh = b_head[0];

    float crit[8];
#pragma unroll
    for (int nt = 0; nt < 8; ++nt) crit[nt] = 0.0f;

    for (int i = 0; i < 4; ++i) {
        if (i < 3) {  // issue next tile into the other buffer, wait current
            async_copy_tile_padded(gx0 + (unsigned)(i + 1) * 4096u,
                                   lx0 + ((i + 1) & 1) * 1088);
            WAIT_ASYNC("0x8");
        } else {
            WAIT_ASYNC("0x0");
        }

        const int   node0 = node_base + i * 16;
        const float* xb   = &sx[(wave * 2 + (i & 1)) * 1088];

        // opaque per-iteration fragment offsets: forbid hoisting into VGPRs
        unsigned foA = 0u, foC = 8192u;
        asm volatile("" : "+v"(foA), "+v"(foC));

        // ---- build A fragments (f32 in LDS -> f16), rows padded to 68 f32 ----
        v16h A[2];
#pragma unroll
        for (int kf = 0; kf < 2; ++kf) {
            const float* rp = xb + m0 * 68 + kf * 32 + half * 8;
            float4 c0 = *(const float4*)(rp);
            float4 c1 = *(const float4*)(rp + 4);
            float4 c2 = *(const float4*)(rp + 16);
            float4 c3 = *(const float4*)(rp + 20);
            v16h a;
            a[0]  = (_Float16)c0.x; a[1]  = (_Float16)c0.y;
            a[2]  = (_Float16)c0.z; a[3]  = (_Float16)c0.w;
            a[4]  = (_Float16)c1.x; a[5]  = (_Float16)c1.y;
            a[6]  = (_Float16)c1.z; a[7]  = (_Float16)c1.w;
            a[8]  = (_Float16)c2.x; a[9]  = (_Float16)c2.y;
            a[10] = (_Float16)c2.z; a[11] = (_Float16)c2.w;
            a[12] = (_Float16)c3.x; a[13] = (_Float16)c3.y;
            a[14] = (_Float16)c3.z; a[15] = (_Float16)c3.w;
            A[kf] = a;
        }

        // ---- actor GEMM: h_a = relu(x @ W_a + b_a) ----
        v8f acc[8];
#pragma unroll
        for (int nt = 0; nt < 8; ++nt) {
            v16h b0 = *(const v16h*)&sfrag[foA + nt * 1024 + lane * 16];
            v16h b1 = *(const v16h*)&sfrag[foA + nt * 1024 + 512 + lane * 16];
            v8f c = {0.f, 0.f, 0.f, 0.f, 0.f, 0.f, 0.f, 0.f};
            c = __builtin_amdgcn_wmma_f32_16x16x32_f16(
                    false, A[0], false, b0, (short)0, c, false, false);
            c = __builtin_amdgcn_wmma_f32_16x16x32_f16(
                    false, A[1], false, b1, (short)0, c, false, false);
#pragma unroll
            for (int r = 0; r < 8; ++r) {
                float h = c[r] + bav[nt];
                c[r] = h > 0.f ? h : 0.f;
            }
            acc[nt] = c;
        }

        // ---- LayerNorm over hidden (row reductions across 16 lanes) ----
        float s1[8], s2[8];
#pragma unroll
        for (int r = 0; r < 8; ++r) { s1[r] = 0.f; s2[r] = 0.f; }
#pragma unroll
        for (int nt = 0; nt < 8; ++nt)
#pragma unroll
            for (int r = 0; r < 8; ++r) {
                float h = acc[nt][r];
                s1[r] += h;
                s2[r] += h * h;
            }
#pragma unroll
        for (int m = 1; m < 16; m <<= 1)
#pragma unroll
            for (int r = 0; r < 8; ++r) {
                s1[r] += __shfl_xor(s1[r], m, 32);
                s2[r] += __shfl_xor(s2[r], m, 32);
            }
        float mu[8], rstd[8];
#pragma unroll
        for (int r = 0; r < 8; ++r) {
            mu[r] = s1[r] * (1.0f / HIDDEN);
            float var = s2[r] * (1.0f / HIDDEN) - mu[r] * mu[r];
            rstd[r] = rsqrtf(var + 1e-5f);
        }

        // ---- scores = LN(h_a) . w_head + b_head ----
        float p[8];
#pragma unroll
        for (int r = 0; r < 8; ++r) p[r] = 0.f;
#pragma unroll
        for (int nt = 0; nt < 8; ++nt)
#pragma unroll
            for (int r = 0; r < 8; ++r) {
                float hn = (acc[nt][r] - mu[r]) * rstd[r] * gv[nt] + bv[nt];
                p[r] += hn * wv[nt];
            }
#pragma unroll
        for (int m = 1; m < 16; m <<= 1)
#pragma unroll
            for (int r = 0; r < 8; ++r) p[r] += __shfl_xor(p[r], m, 32);
#pragma unroll
        for (int r = 0; r < 8; ++r)
            if (m0 == r)
                scores[(size_t)g * N_MAX + node0 + r + half * 8] = p[r] + bh;

        // ---- critic GEMM: masked column sums of relu(x @ W_c + b_c) ----
        float mval = (float)node_mask[g * N_MAX + node0 + m0];
        float mrow[8];
#pragma unroll
        for (int r = 0; r < 8; ++r) mrow[r] = __shfl(mval, r + half * 8, 32);
#pragma unroll
        for (int nt = 0; nt < 8; ++nt) {
            v16h b0 = *(const v16h*)&sfrag[foC + nt * 1024 + lane * 16];
            v16h b1 = *(const v16h*)&sfrag[foC + nt * 1024 + 512 + lane * 16];
            v8f c = {0.f, 0.f, 0.f, 0.f, 0.f, 0.f, 0.f, 0.f};
            c = __builtin_amdgcn_wmma_f32_16x16x32_f16(
                    false, A[0], false, b0, (short)0, c, false, false);
            c = __builtin_amdgcn_wmma_f32_16x16x32_f16(
                    false, A[1], false, b1, (short)0, c, false, false);
#pragma unroll
            for (int r = 0; r < 8; ++r) {
                float h = c[r] + bcv[nt];
                h = h > 0.f ? h : 0.f;
                crit[nt] += h * mrow[r];
            }
        }
    }

    // block-level reduction of critic partial column sums, then one atomic set
#pragma unroll
    for (int nt = 0; nt < 8; ++nt)
        atomicAdd(&lds_c[nt * 16 + m0], crit[nt]);
    __syncthreads();
    if (threadIdx.x < HIDDEN)
        atomicAdd(&E_sum[g * HIDDEN + threadIdx.x], lds_c[threadIdx.x]);
}

// =====================================================================
// Kernel 4: critic finalize (1 block, 128 threads)
// =====================================================================
__global__ __launch_bounds__(128) void finalize_kernel(
    const float* __restrict__ E_sum, const float* __restrict__ cnt,
    const float* __restrict__ w_attn, const float* __restrict__ b_attn,
    const float* __restrict__ Wc1, const float* __restrict__ bc1,
    const float* __restrict__ Wc2, const float* __restrict__ bc2,
    float* __restrict__ out_v)
{
    __shared__ float E[R_GRAPHS * HIDDEN];
    __shared__ float a[R_GRAPHS];
    __shared__ float w[R_GRAPHS];
    __shared__ float gvec[HIDDEN];
    __shared__ float yvec[HIDDEN];

    int tid = threadIdx.x;
    for (int r = 0; r < R_GRAPHS; ++r)
        E[r * HIDDEN + tid] = E_sum[r * HIDDEN + tid] / cnt[r];
    __syncthreads();

    if (tid < R_GRAPHS) {
        float s = 0.f;
        for (int h = 0; h < HIDDEN; ++h) s += E[tid * HIDDEN + h] * w_attn[h];
        a[tid] = s + b_attn[0];
    }
    __syncthreads();

    if (tid == 0) {
        float amax = a[0];
        for (int r = 1; r < R_GRAPHS; ++r) amax = fmaxf(amax, a[r]);
        float ssum = 0.f;
        for (int r = 0; r < R_GRAPHS; ++r) {
            float e = expf(a[r] - amax);
            w[r] = e;
            ssum += e;
        }
        float inv = 1.0f / ssum;
        for (int r = 0; r < R_GRAPHS; ++r) w[r] *= inv;
    }
    __syncthreads();

    {
        float s = 0.f;
        for (int r = 0; r < R_GRAPHS; ++r) s += w[r] * E[r * HIDDEN + tid];
        gvec[tid] = s;
    }
    __syncthreads();

    {
        float s = bc1[tid];
        for (int k = 0; k < HIDDEN; ++k) s += gvec[k] * Wc1[k * HIDDEN + tid];
        yvec[tid] = (s > 0.f ? s : 0.f) * Wc2[tid];
    }
    __syncthreads();

    for (int off = 64; off > 0; off >>= 1) {
        if (tid < off) yvec[tid] += yvec[tid + off];
        __syncthreads();
    }
    if (tid == 0) out_v[0] = yvec[0] + bc2[0];
}

// =====================================================================
// Kernel 5: candidate logits (gather + stable compaction). grid 64 x 512.
// =====================================================================
__global__ __launch_bounds__(512) void logits_kernel(
    const int* __restrict__ cand_idx, const int* __restrict__ cand_mask,
    const int* __restrict__ node_mask, const float* __restrict__ scores,
    float* __restrict__ out)
{
    int r = blockIdx.x;
    int j = threadIdx.x;

    int idx = cand_idx[r * K_MAX + j];
    int ok  = (cand_mask[r * K_MAX + j] != 0) &&
              (node_mask[r * N_MAX + idx] != 0);
    float val = NEGV;
    if (ok) val = 5.0f * tanhf(scores[(size_t)r * N_MAX + idx]);

    __shared__ int   sc[K_MAX];
    __shared__ float outs[K_MAX];
    sc[j]   = ok;
    outs[j] = NEGV;
    __syncthreads();

    for (int off = 1; off < K_MAX; off <<= 1) {
        int t = (j >= off) ? sc[j - off] : 0;
        __syncthreads();
        sc[j] += t;
        __syncthreads();
    }
    if (ok) outs[sc[j] - 1] = val;
    __syncthreads();
    out[(size_t)r * K_MAX + j] = outs[j];
}

// =====================================================================
extern "C" void kernel_launch(void* const* d_in, const int* in_sizes, int n_in,
                              void* d_out, int out_size, void* d_ws, size_t ws_size,
                              hipStream_t stream) {
    (void)in_sizes; (void)n_in; (void)out_size; (void)ws_size;

    const float* x         = (const float*)d_in[0];
    const int*   node_mask = (const int*)  d_in[1];
    const int*   cand_idx  = (const int*)  d_in[2];
    const int*   cand_mask = (const int*)  d_in[3];
    const float* W_a       = (const float*)d_in[4];
    const float* b_a       = (const float*)d_in[5];
    const float* W_c       = (const float*)d_in[6];
    const float* b_c       = (const float*)d_in[7];
    const float* ln_g      = (const float*)d_in[8];
    const float* ln_b      = (const float*)d_in[9];
    const float* w_head    = (const float*)d_in[10];
    const float* b_head    = (const float*)d_in[11];
    const float* Wc1       = (const float*)d_in[12];
    const float* bc1       = (const float*)d_in[13];
    const float* Wc2       = (const float*)d_in[14];
    const float* bc2       = (const float*)d_in[15];
    const float* w_attn    = (const float*)d_in[16];
    const float* b_attn    = (const float*)d_in[17];

    char* ws = (char*)d_ws;
    _Float16* fragA  = (_Float16*)ws;                  // 8192 f16
    _Float16* fragC  = (_Float16*)(ws + 16384);        // 8192 f16 (contiguous!)
    float*    E_sum  = (float*)(ws + 32768);           // 64*128 f32
    float*    cnt    = (float*)(ws + 65536);           // 64 f32
    float*    scores = (float*)(ws + 65792);           // 64*4096 f32

    float* out_logits = (float*)d_out;                 // 64*512
    float* out_v      = out_logits + R_GRAPHS * K_MAX; // scalar

    prep_kernel<<<dim3(64), dim3(256), 0, stream>>>(W_a, W_c, fragA, fragC, E_sum);
    count_kernel<<<dim3(R_GRAPHS), dim3(256), 0, stream>>>(node_mask, cnt);
    encode_kernel<<<dim3(8, R_GRAPHS), dim3(256), 0, stream>>>(
        x, node_mask, b_a, b_c, ln_g, ln_b, w_head, b_head,
        fragA, E_sum, scores);
    finalize_kernel<<<dim3(1), dim3(128), 0, stream>>>(
        E_sum, cnt, w_attn, b_attn, Wc1, bc1, Wc2, bc2, out_v);
    logits_kernel<<<dim3(R_GRAPHS), dim3(K_MAX), 0, stream>>>(
        cand_idx, cand_mask, node_mask, scores, out_logits);
}
